// MultiHeadAttention_11888469475693
// MI455X (gfx1250) — matrix operand order repaired and checked by reading, not compiled, not run on hardware
//
#include <hip/hip_runtime.h>

// ---------------------------------------------------------------------------
// MI455X (gfx1250, wave32) multi-head causal attention with RoPE.
// Compute-bound (~51 GFLOP vs ~35MB I/O @ 23.3 TB/s) -> everything on the
// f16 WMMA path: v_wmma_f32_16x16x32_f16 (16x16 f32 acc = 8 VGPRs/lane).
// Per wave: 32x64 output slab (8 accumulators) to maximize WMMA density per
// LDS byte; double-buffered LDS staging + global_prefetch_b8 for latency.
// ---------------------------------------------------------------------------

#define Bn 4
#define Tn 2048
#define Cn 512
#define Hn 8
#define Dn 64
#define F3 1536   // 3*C

typedef __attribute__((ext_vector_type(16))) _Float16 v16h;
typedef __attribute__((ext_vector_type(8)))  float    v8f;
typedef __attribute__((ext_vector_type(2)))  _Float16 h2;

__device__ __forceinline__ v8f wmma_f16(v16h a, v16h b, v8f c) {
  return __builtin_amdgcn_wmma_f32_16x16x32_f16(false, a, false, b,
                                                (short)0, c, false, false);
}

// A-fragment (16x32 f16, MxK): lane m = lane&15; element pair j lives at
// k = (lane>>4)*16 + 2*j   (contiguous, matches B fragments)
__device__ __forceinline__ v16h load_a_frag_f32(const float* row, int hl) {
  v16h a;
#pragma unroll
  for (int j = 0; j < 8; ++j) {
    int k = hl * 16 + (j << 1);
    float2 p = *(const float2*)(row + k);
    a[2 * j]     = (_Float16)p.x;
    a[2 * j + 1] = (_Float16)p.y;
  }
  return a;
}

__device__ __forceinline__ v16h load_a_frag_f16(const _Float16* row, int hl) {
  v16h a;
#pragma unroll
  for (int j = 0; j < 8; ++j) {
    int k = hl * 16 + (j << 1);
    h2 p = *(const h2*)(row + k);
    a[2 * j]     = p.x;
    a[2 * j + 1] = p.y;
  }
  return a;
}

// ---------------------------------------------------------------------------
// Kernel 1: qkv = x @ Wqkv. Block tile 128x64 (4 waves x 32 rows), fused RoPE.
// Each 64-col tile is exactly one (head, {q|k|v}) slab since 3*D = 3*64.
// RoPE couples d and d+32 (same angle) -> accumulator pairs (0,2),(1,3).
// ---------------------------------------------------------------------------
__global__ void __launch_bounds__(128) qkv_rope_kernel(
    const float* __restrict__ x, const float* __restrict__ Wqkv,
    _Float16* __restrict__ Q, _Float16* __restrict__ K,
    _Float16* __restrict__ V) {
  __shared__ _Float16 bt[2][64 * 32];  // W tile transposed [f_local][k], dbuf

  const int lane = threadIdx.x & 31, wave = threadIdx.x >> 5;
  const int hl = lane >> 4, ln = lane & 15;
  const int m0 = blockIdx.x * 128;       // row block (rows of B*T)
  const int nblk = blockIdx.y;           // 0..23
  const int f0 = nblk * 64;
  const int h = nblk / 3, part = nblk % 3;

  auto stage = [&](int buf, int k0) {    // 32x64 W tile -> LDS transposed
#pragma unroll
    for (int pass = 0; pass < 4; ++pass) {
      int k = (threadIdx.x >> 4) + pass * 8;
      int f = (threadIdx.x & 15) * 4;
      float4 w = *(const float4*)(Wqkv + (size_t)(k0 + k) * F3 + f0 + f);
      bt[buf][(f + 0) * 32 + k] = (_Float16)w.x;
      bt[buf][(f + 1) * 32 + k] = (_Float16)w.y;
      bt[buf][(f + 2) * 32 + k] = (_Float16)w.z;
      bt[buf][(f + 3) * 32 + k] = (_Float16)w.w;
    }
  };

  v8f acc[2][4] = {};
  const float* xrow0 = x + (size_t)(m0 + wave * 32 + ln) * Cn;
  const float* xrow1 = xrow0 + (size_t)16 * Cn;

  stage(0, 0);
  __syncthreads();
  for (int k0 = 0; k0 < Cn; k0 += 32) {
    int cur = (k0 >> 5) & 1;
    if (k0 + 32 < Cn) stage(cur ^ 1, k0 + 32);          // overlap with WMMAs
    if (k0 + 64 < Cn) {
      __builtin_prefetch(xrow0 + k0 + 64, 0, 3);        // global_prefetch_b8
      __builtin_prefetch(xrow1 + k0 + 64, 0, 3);
    }
    v16h af0 = load_a_frag_f32(xrow0 + k0, hl);
    v16h af1 = load_a_frag_f32(xrow1 + k0, hl);
#pragma unroll
    for (int nt = 0; nt < 4; ++nt) {
      // B-fragment (32x16): lane n = nt*16+ln, k = hl*16 + e (contiguous)
      v16h bf = *(const v16h*)(&bt[cur][(nt * 16 + ln) * 32 + hl * 16]);
      acc[0][nt] = wmma_f16(af0, bf, acc[0][nt]);
      acc[1][nt] = wmma_f16(af1, bf, acc[1][nt]);
    }
    __syncthreads();
  }

#pragma unroll
  for (int rg = 0; rg < 2; ++rg) {
    const int mb = m0 + wave * 32 + rg * 16;
    if (part == 2) {  // V: no RoPE
#pragma unroll
      for (int nt = 0; nt < 4; ++nt)
#pragma unroll
        for (int r = 0; r < 8; ++r) {
          int m = mb + r + 8 * hl;
          int b = m >> 11, t = m & (Tn - 1);
          V[((size_t)(b * Hn + h) * Tn + t) * Dn + nt * 16 + ln] =
              (_Float16)acc[rg][nt][r];
        }
    } else {
      _Float16* dst = (part == 0) ? Q : K;
#pragma unroll
      for (int pr = 0; pr < 2; ++pr) {   // pairs (0,2),(1,3): d and d+32
        int dmod = pr * 16 + ln;         // d % 32
        float invf = __powf(10000.f, -(float)dmod * (1.0f / 32.0f));
#pragma unroll
        for (int r = 0; r < 8; ++r) {
          int m = mb + r + 8 * hl;
          int b = m >> 11, t = m & (Tn - 1);
          float sn, cs;
          __sincosf((float)t * invf, &sn, &cs);
          float vlo = acc[rg][pr][r], vhi = acc[rg][pr + 2][r];
          size_t base = ((size_t)(b * Hn + h) * Tn + t) * Dn;
          dst[base + dmod]      = (_Float16)(vlo * cs - vhi * sn);
          dst[base + 32 + dmod] = (_Float16)(vhi * cs + vlo * sn);
        }
      }
    }
  }
}

// ---------------------------------------------------------------------------
// Kernel 2: causal flash attention. Block = (b,h) x 64 query rows; each wave
// owns 16 rows. 32 keys/step: S via 2x2 WMMAs (K frags = contiguous 32B
// global loads), online softmax with half-wave shfl reductions, P -> LDS
// round trip into A-layout, O += P@V with double-buffered transposed V tile.
// ---------------------------------------------------------------------------
__global__ void __launch_bounds__(128) attn_kernel(
    const _Float16* __restrict__ Q, const _Float16* __restrict__ K,
    const _Float16* __restrict__ V, _Float16* __restrict__ AO) {
  __shared__ _Float16 vt[2][64 * 32];    // V tile transposed [d][key], dbuf
  __shared__ _Float16 pbuf[4][16 * 32];  // per-wave P scratch [m][k]

  const int lane = threadIdx.x & 31, wave = threadIdx.x >> 5;
  const int hl = lane >> 4, ln = lane & 15;
  const int q0 = blockIdx.x * 64;
  const int bh = blockIdx.y;
  const int b = bh >> 3, h = bh & 7;

  const _Float16* Qb = Q + (size_t)bh * Tn * Dn;
  const _Float16* Kb = K + (size_t)bh * Tn * Dn;
  const _Float16* Vb = V + (size_t)bh * Tn * Dn;

  auto stageV = [&](int buf, int kk) {   // 32 keys x 64 d -> LDS transposed
    int key = threadIdx.x >> 2;
    int d0 = (threadIdx.x & 3) * 16;
    const _Float16* vp = Vb + (size_t)(kk + key) * Dn + d0;
#pragma unroll
    for (int d = 0; d < 16; ++d) vt[buf][(d0 + d) * 32 + key] = vp[d];
  };

  // Q fragments: rows q0+wave*16+{0..15}, K-dim split d:[0,32),[32,64)
  const _Float16* qrow = Qb + (size_t)(q0 + wave * 16 + ln) * Dn;
  v16h qf0 = load_a_frag_f16(qrow, hl);
  v16h qf1 = load_a_frag_f16(qrow + 32, hl);

  v8f o[4] = {};
  float rm[8], rl[8];
#pragma unroll
  for (int r = 0; r < 8; ++r) { rm[r] = -1e30f; rl[r] = 0.f; }

  const int nsteps = q0 / 32 + 2;  // covers keys <= q0+63 (q0 <= 1984)
  stageV(0, 0);
  __syncthreads();
  for (int step = 0; step < nsteps; ++step) {
    const int kk = step * 32;
    const int cur = step & 1;
    if (step + 1 < nsteps) stageV(cur ^ 1, kk + 32);    // overlap staging
    if (step + 2 < nsteps) {                            // L2 prefetch, 2 ahead
      __builtin_prefetch(Kb + (size_t)(kk + 64 + lane) * Dn, 0, 3);
      __builtin_prefetch(Vb + (size_t)(kk + 64 + lane) * Dn, 0, 3);
    }

    // S = Q @ K^T : two 16-key tiles
    float sv[2][8];
#pragma unroll
    for (int ht = 0; ht < 2; ++ht) {
      int kbase = kk + ht * 16;
      // K fragment: lane key = kbase+ln, d = f*32 + hl*16 + e -> 32B contig.
      v8f s = {};
      v16h kf0 = *(const v16h*)(Kb + (size_t)(kbase + ln) * Dn + hl * 16);
      v16h kf1 = *(const v16h*)(Kb + (size_t)(kbase + ln) * Dn + 32 + hl * 16);
      s = wmma_f16(qf0, kf0, s);
      s = wmma_f16(qf1, kf1, s);
#pragma unroll
      for (int r = 0; r < 8; ++r) sv[ht][r] = s[r];
    }

    // causal mask + scale + running row max
    const float scale = 0.125f;  // 1/sqrt(64)
    float tmax[8];
#pragma unroll
    for (int r = 0; r < 8; ++r) {
      int qi = q0 + wave * 16 + r + 8 * hl;
      float s0 = (kk + ln <= qi)      ? sv[0][r] * scale : -3e38f;
      float s1 = (kk + 16 + ln <= qi) ? sv[1][r] * scale : -3e38f;
      sv[0][r] = s0; sv[1][r] = s1;
      tmax[r] = fmaxf(s0, s1);
    }
#pragma unroll
    for (int m = 1; m < 16; m <<= 1)
#pragma unroll
      for (int r = 0; r < 8; ++r)
        tmax[r] = fmaxf(tmax[r], __shfl_xor(tmax[r], m, 32));

    float alpha[8];
#pragma unroll
    for (int r = 0; r < 8; ++r) {
      float mnew = fmaxf(rm[r], tmax[r]);
      alpha[r] = __expf(rm[r] - mnew);
      rm[r] = mnew;
    }

    // P = exp(S - m), stash into per-wave LDS in [m][k] order
    _Float16* pb = &pbuf[wave][0];
    float rs[8];
#pragma unroll
    for (int r = 0; r < 8; ++r) {
      float p0 = __expf(sv[0][r] - rm[r]);
      float p1 = __expf(sv[1][r] - rm[r]);
      rs[r] = p0 + p1;
      int row = r + 8 * hl;
      pb[row * 32 + ln]      = (_Float16)p0;
      pb[row * 32 + 16 + ln] = (_Float16)p1;
    }
#pragma unroll
    for (int m = 1; m < 16; m <<= 1)
#pragma unroll
      for (int r = 0; r < 8; ++r) rs[r] += __shfl_xor(rs[r], m, 32);
#pragma unroll
    for (int r = 0; r < 8; ++r) rl[r] = rl[r] * alpha[r] + rs[r];

    // rescale running O
#pragma unroll
    for (int nt = 0; nt < 4; ++nt)
#pragma unroll
      for (int r = 0; r < 8; ++r) o[nt][r] *= alpha[r];

    asm volatile("s_wait_dscnt 0" ::: "memory");  // P round-trip RAW
    v16h pf = load_a_frag_f16(pb + ln * 32, hl);  // P as A-fragment (16x32)
#pragma unroll
    for (int nt = 0; nt < 4; ++nt) {
      // V fragment: lane n(d) = nt*16+ln, k(key) = hl*16 + e (contig in vt)
      v16h vf = *(const v16h*)(&vt[cur][(nt * 16 + ln) * 32 + hl * 16]);
      o[nt] = wmma_f16(pf, vf, o[nt]);
    }
    __syncthreads();
  }

  // epilogue: O /= l, write [B,T,C] f16 for the projection GEMM
#pragma unroll
  for (int nt = 0; nt < 4; ++nt)
#pragma unroll
    for (int r = 0; r < 8; ++r) {
      int t = q0 + wave * 16 + r + 8 * hl;
      int c = h * Dn + nt * 16 + ln;
      AO[(size_t)(b * Tn + t) * Cn + c] = (_Float16)(o[nt][r] / rl[r]);
    }
}

// ---------------------------------------------------------------------------
// Kernel 3: out = AO @ Wproj (f16 A, f32 out), 128x64 block tile, dbuf LDS.
// ---------------------------------------------------------------------------
__global__ void __launch_bounds__(128) proj_kernel(
    const _Float16* __restrict__ AO, const float* __restrict__ Wproj,
    float* __restrict__ out) {
  __shared__ _Float16 bt[2][64 * 32];

  const int lane = threadIdx.x & 31, wave = threadIdx.x >> 5;
  const int hl = lane >> 4, ln = lane & 15;
  const int m0 = blockIdx.x * 128;
  const int f0 = blockIdx.y * 64;

  auto stage = [&](int buf, int k0) {
#pragma unroll
    for (int pass = 0; pass < 4; ++pass) {
      int k = (threadIdx.x >> 4) + pass * 8;
      int f = (threadIdx.x & 15) * 4;
      float4 w = *(const float4*)(Wproj + (size_t)(k0 + k) * Cn + f0 + f);
      bt[buf][(f + 0) * 32 + k] = (_Float16)w.x;
      bt[buf][(f + 1) * 32 + k] = (_Float16)w.y;
      bt[buf][(f + 2) * 32 + k] = (_Float16)w.z;
      bt[buf][(f + 3) * 32 + k] = (_Float16)w.w;
    }
  };

  v8f acc[2][4] = {};
  const _Float16* arow0 = AO + (size_t)(m0 + wave * 32 + ln) * Cn;
  const _Float16* arow1 = arow0 + (size_t)16 * Cn;

  stage(0, 0);
  __syncthreads();
  for (int k0 = 0; k0 < Cn; k0 += 32) {
    int cur = (k0 >> 5) & 1;
    if (k0 + 32 < Cn) stage(cur ^ 1, k0 + 32);
    if (k0 + 64 < Cn) {
      __builtin_prefetch(arow0 + k0 + 64, 0, 3);
      __builtin_prefetch(arow1 + k0 + 64, 0, 3);
    }
    v16h af0 = load_a_frag_f16(arow0 + k0, hl);
    v16h af1 = load_a_frag_f16(arow1 + k0, hl);
#pragma unroll
    for (int nt = 0; nt < 4; ++nt) {
      v16h bf = *(const v16h*)(&bt[cur][(nt * 16 + ln) * 32 + hl * 16]);
      acc[0][nt] = wmma_f16(af0, bf, acc[0][nt]);
      acc[1][nt] = wmma_f16(af1, bf, acc[1][nt]);
    }
    __syncthreads();
  }

#pragma unroll
  for (int rg = 0; rg < 2; ++rg)
#pragma unroll
    for (int nt = 0; nt < 4; ++nt)
#pragma unroll
      for (int r = 0; r < 8; ++r) {
        int m = m0 + wave * 32 + rg * 16 + r + 8 * hl;
        out[(size_t)m * Cn + f0 + nt * 16 + ln] = acc[rg][nt][r];
      }
}

// ---------------------------------------------------------------------------
extern "C" void kernel_launch(void* const* d_in, const int* in_sizes, int n_in,
                              void* d_out, int out_size, void* d_ws,
                              size_t ws_size, hipStream_t stream) {
  const float* x     = (const float*)d_in[0];
  const float* Wqkv  = (const float*)d_in[1];
  const float* Wproj = (const float*)d_in[2];
  float* out = (float*)d_out;

  // Workspace: Q,K,V [B,H,T,D] f16 (8 MiB each) + AO [B,T,C] f16 (8 MiB).
  const size_t SLAB = (size_t)Bn * Hn * Tn * Dn * sizeof(_Float16);  // 8 MiB
  char* ws = (char*)d_ws;
  _Float16* Q  = (_Float16*)(ws);
  _Float16* K  = (_Float16*)(ws + SLAB);
  _Float16* V  = (_Float16*)(ws + 2 * SLAB);
  _Float16* AO = (_Float16*)(ws + 3 * SLAB);

  qkv_rope_kernel<<<dim3((Bn * Tn) / 128, F3 / 64), 128, 0, stream>>>(
      x, Wqkv, Q, K, V);
  attn_kernel<<<dim3(Tn / 64, Bn * Hn), 128, 0, stream>>>(Q, K, V, AO);
  proj_kernel<<<dim3((Bn * Tn) / 128, Cn / 64), 128, 0, stream>>>(
      AO, Wproj, out);
}
